// CrossBandBiMamba_28819230556259
// MI455X (gfx1250) — compile-verified, matrix operand order checked by reading, not run on hardware
//
#include <hip/hip_runtime.h>
#include <hip/hip_bf16.h>
#include <math.h>

// ---------------- problem constants ----------------
#define CH      128
#define DSTATE  16
#define LBAND   30          // sequence length for the mamba scan (K bands)
#define BB_     8
#define TT_     300
#define SEQN    (BB_ * TT_)         // 2400 sequences
#define TOK     (SEQN * LBAND)      // 72000 tokens
#define MTILES  (TOK / 16)          // 4500

typedef __attribute__((ext_vector_type(16))) __bf16 v16bf;
typedef __attribute__((ext_vector_type(8)))  float  v8f;
typedef __attribute__((ext_vector_type(4)))  int    vi4;

#define HAS_ASYNC_LDS (__has_builtin(__builtin_amdgcn_global_load_async_to_lds_b128) && \
                       __has_builtin(__builtin_amdgcn_s_wait_asynccnt))

// ---------------- scalar helpers ----------------
__device__ __forceinline__ unsigned short f2bf(float f) {
    unsigned int u = __float_as_uint(f);
    unsigned int r = u + 0x7FFFu + ((u >> 16) & 1u);
    return (unsigned short)(r >> 16);
}
__device__ __forceinline__ float bf2f(unsigned short h) {
    return __uint_as_float(((unsigned int)h) << 16);
}
__device__ __forceinline__ float fast_rcp(float x) {
#if __has_builtin(__builtin_amdgcn_rcpf)
    return __builtin_amdgcn_rcpf(x);
#else
    return 1.f / x;
#endif
}
__device__ __forceinline__ float sigmoid_f(float v) {
    return fast_rcp(1.f + __expf(-v));        // v_exp_f32 + v_rcp_f32, no IEEE div
}
__device__ __forceinline__ float silu_f(float v) { return v * sigmoid_f(v); }

// row index mapping for the flipped (backward) direction
__device__ __forceinline__ int maprow(int r, int flip) {
    if (!flip) return r;
    int seq = r / LBAND;
    int l   = r - seq * LBAND;
    return seq * LBAND + (LBAND - 1 - l);
}

// ---------------- async global -> LDS staging (whole weight matrix per block) ----
// 256 threads; nbytes must be a multiple of 16.
__device__ __forceinline__ void stage_to_lds(const unsigned short* gsrc,
                                             unsigned short* lds, int nbytes) {
#if HAS_ASYNC_LDS
    char* gp = (char*)const_cast<unsigned short*>(gsrc);
    char* lp = (char*)lds;
    for (int i = threadIdx.x * 16; i < nbytes; i += 256 * 16) {
        __builtin_amdgcn_global_load_async_to_lds_b128(
            (__attribute__((address_space(1))) vi4*)(gp + i),
            (__attribute__((address_space(3))) vi4*)(lp + i),
            0, 0);
    }
    __builtin_amdgcn_s_wait_asynccnt(0);   // each wave waits for its own copies
    __syncthreads();                       // then all waves have arrived -> LDS complete
#else
    for (int i = threadIdx.x * 16; i < nbytes; i += 256 * 16) {
        *reinterpret_cast<int4*>((char*)lds + i) =
            *reinterpret_cast<const int4*>((const char*)gsrc + i);
    }
    __syncthreads();
#endif
}

// ---------------- WMMA tile loader (CDNA5 16-bit A layout; B uses same via N-major) --
// rowbase: per-lane pointer to row start (row = tile_row + lane%16);
// element pair e2 -> k = k0 + 2*e2 + 8*g + (e2>=4 ? 8 : 0), g = lane>>4.
__device__ __forceinline__ v16bf load_tile(const unsigned short* rowbase, int k0, int g) {
    union { v16bf v; unsigned int u[8]; } r;
#pragma unroll
    for (int e2 = 0; e2 < 8; ++e2) {
        int k = k0 + 2 * e2 + 8 * g + ((e2 >= 4) ? 8 : 0);
        r.u[e2] = *reinterpret_cast<const unsigned int*>(rowbase + k);
    }
    return r.v;
}

__device__ __forceinline__ v8f wmma_bf16(v16bf a, v16bf b, v8f c) {
    return __builtin_amdgcn_wmma_f32_16x16x32_bf16(
        false, a, false, b, (short)0, c, false, false);
}

// ---------------- setup: fold + transpose weights to bf16 ----------------
// grid 256 blocks (dir*128 + i), 128 threads (j)
__global__ void setup_weights(
    const float* f_Win, const float* f_Wx, const float* f_Wdt, const float* f_Alog,
    const float* f_Wout, const float* f_bout,
    const float* b_Win, const float* b_Wx, const float* b_Wdt, const float* b_Alog,
    const float* b_Wout, const float* b_bout,
    const float* comb_W, const float* comb_b,
    unsigned short* WinT, unsigned short* WeffT, unsigned short* WcombT,
    float* Amat, float* biasTot)
{
    int bid = blockIdx.x;
    int dir = bid >> 7;
    int i   = bid & 127;         // weight row (k index)
    int j   = threadIdx.x;       // 0..127

    const float* Win  = dir ? b_Win  : f_Win;
    const float* Wx   = dir ? b_Wx   : f_Wx;
    const float* Wdt  = dir ? b_Wdt  : f_Wdt;
    const float* Alog = dir ? b_Alog : f_Alog;
    const float* Wout = dir ? b_Wout : f_Wout;

    unsigned short* winT = WinT  + dir * 256 * 128;
    unsigned short* weT  = WeffT + dir * 160 * 128;
    unsigned short* wcT  = WcombT + dir * 128 * 128;

    // WinT[n][i] = Win[i][n]  (n = j and j+128)
    winT[j * 128 + i]         = f2bf(Win[i * 256 + j]);
    winT[(j + 128) * 128 + i] = f2bf(Win[i * 256 + 128 + j]);

    // Weff cols 0..31 are Wx cols 8..39 (B then C): WeffT[n][i] = Wx[i][8+n]
    if (j < 32) weT[j * 128 + i] = f2bf(Wx[i * 40 + 8 + j]);

    // dt fold: WeffT[32+j][i] = sum_r Wx[i][r] * Wdt[r][j]
    float acc = 0.f;
#pragma unroll
    for (int r = 0; r < 8; ++r) acc += Wx[i * 40 + r] * Wdt[r * 128 + j];
    weT[(32 + j) * 128 + i] = f2bf(acc);

    // comb fold: WcombT[j][i] = sum_c Wout[i][c] * comb_W[dir*128+c][j]
    float wc = 0.f;
    for (int c = 0; c < 128; ++c)
        wc += Wout[i * 128 + c] * comb_W[(dir * 128 + c) * 128 + j];
    wcT[j * 128 + i] = f2bf(wc);

    // A = -exp(Alog)
    if (j < DSTATE) Amat[dir * 128 * DSTATE + i * DSTATE + j] = -__expf(Alog[i * DSTATE + j]);

    // folded output bias (once)
    if (bid == 0) {
        float bt = comb_b[j];
        for (int c = 0; c < 128; ++c)
            bt += f_bout[c] * comb_W[c * 128 + j]
                + b_bout[c] * comb_W[(128 + c) * 128 + j];
        biasTot[j] = bt;
    }
}

// ---------------- layernorm over channels (wave32 per token) ----------------
__global__ void __launch_bounds__(256) ln_kernel(
    const float* __restrict__ x, const float* g, const float* beta,
    unsigned short* __restrict__ xn)
{
    int warp = threadIdx.x >> 5;
    int lane = threadIdx.x & 31;
    int tk = blockIdx.x * 8 + warp;
    if (tk >= TOK) return;
    int k  = tk % LBAND;
    int bt = tk / LBAND;
    int t  = bt % TT_;
    int b  = bt / TT_;
    const float* base = x + (long)b * CH * TT_ * LBAND + (long)t * LBAND + k;
    float v[4], s = 0.f, s2 = 0.f;
#pragma unroll
    for (int j = 0; j < 4; ++j) {
        int n = lane + 32 * j;
        v[j] = base[(long)n * (TT_ * LBAND)];
        s += v[j]; s2 += v[j] * v[j];
    }
#pragma unroll
    for (int m = 16; m; m >>= 1) {
        s  += __shfl_xor(s,  m, 32);
        s2 += __shfl_xor(s2, m, 32);
    }
    float mu  = s * (1.f / 128.f);
    float var = s2 * (1.f / 128.f) - mu * mu;
    float rs  = rsqrtf(var + 1e-5f);
    unsigned short* o = xn + (long)tk * 128;
#pragma unroll
    for (int j = 0; j < 4; ++j) {
        int n = lane + 32 * j;
        o[n] = f2bf((v[j] - mu) * rs * g[n] + beta[n]);
    }
}

// ---------------- GEMM: xz = xn @ Win + bin ; split x / silu(z) ----------------
// 1 block / 16-row M-tile; weights (256x128 bf16 = 64KB) staged in LDS; 8 wmma/wave.
__global__ void __launch_bounds__(256) gemm_in(
    const unsigned short* __restrict__ xn,
    const unsigned short* __restrict__ winT,
    const float* __restrict__ bin,
    unsigned short* __restrict__ xzx,
    unsigned short* __restrict__ sz, int flip)
{
    __shared__ __align__(16) unsigned short lw[256 * 128];
    stage_to_lds(winT, lw, 256 * 128 * 2);

    int wv = threadIdx.x >> 5, lane = threadIdx.x & 31;
    int g = lane >> 4, lr = lane & 15;
    int m0 = blockIdx.x * 16;
    const unsigned short* arow  = xn + (long)maprow(m0 + lr, flip) * 128;
    const unsigned short* b0row = lw + (wv * 16 + lr) * 128;         // n-tile wv   (x half)
    const unsigned short* b1row = lw + ((wv + 8) * 16 + lr) * 128;   // n-tile wv+8 (z half)
    v8f c0 = {}, c1 = {};
#pragma unroll
    for (int k0 = 0; k0 < 128; k0 += 32) {
        v16bf a = load_tile(arow, k0, g);             // A regs reused for both N-tiles
        c0 = wmma_bf16(a, load_tile(b0row, k0, g), c0);
        c1 = wmma_bf16(a, load_tile(b1row, k0, g), c1);
    }
    int nx = wv * 16 + lr;
    float bx = bin[nx];
    float bz = bin[128 + nx];
#pragma unroll
    for (int r = 0; r < 8; ++r) {
        int m = m0 + r + 8 * g;
        xzx[(long)m * 128 + nx] = f2bf(c0[r] + bx);
        float zv = c1[r] + bz;
        sz[(long)m * 128 + nx] = f2bf(zv * sigmoid_f(zv));
    }
}

// ---------------- depthwise causal conv (len 4) + silu ----------------
__global__ void __launch_bounds__(256) conv_silu(
    const unsigned short* __restrict__ xzx,
    const float* __restrict__ convw, const float* __restrict__ convb,
    unsigned short* __restrict__ xp)
{
    int gid = blockIdx.x * 256 + threadIdx.x;     // SEQN*128 = 307200
    int seq = gid >> 7, d = gid & 127;
    float w0 = convw[d * 4 + 0], w1 = convw[d * 4 + 1];
    float w2 = convw[d * 4 + 2], w3 = convw[d * 4 + 3];
    float cb = convb[d];
    float x0 = 0.f, x1 = 0.f, x2 = 0.f, x3 = 0.f;
    const unsigned short* in  = xzx + (long)seq * LBAND * 128 + d;
    unsigned short*       out = xp  + (long)seq * LBAND * 128 + d;
    for (int l = 0; l < LBAND; ++l) {
        x0 = x1; x1 = x2; x2 = x3;
        x3 = bf2f(in[l * 128]);
        float v = cb + w0 * x0 + w1 * x1 + w2 * x2 + w3 * x3;
        out[l * 128] = f2bf(silu_f(v));
    }
}

// ---------------- GEMM: [B | C | dt] = xp @ Weff ; softplus(dt+bdt) ----------------
// 1 block / M-tile; Weff (160x128 bf16 = 40KB) staged in LDS; A regs preloaded once.
__global__ void __launch_bounds__(256) gemm_x(
    const unsigned short* __restrict__ xp,
    const unsigned short* __restrict__ weT,
    const float* __restrict__ bdt,
    float* __restrict__ Bbuf, float* __restrict__ Cbuf,
    float* __restrict__ dtb)
{
    __shared__ __align__(16) unsigned short lw[160 * 128];
    stage_to_lds(weT, lw, 160 * 128 * 2);

    int wv = threadIdx.x >> 5, lane = threadIdx.x & 31;
    int g = lane >> 4, lr = lane & 15;
    int m0 = blockIdx.x * 16;
    const unsigned short* arow = xp + (long)(m0 + lr) * 128;
    v16bf a[4];
#pragma unroll
    for (int i = 0; i < 4; ++i) a[i] = load_tile(arow, i * 32, g);

    for (int nt = wv; nt < 10; nt += 8) {        // waves 0..7 get nt=wv; waves 0,1 also nt+8
        const unsigned short* brow = lw + (nt * 16 + lr) * 128;
        v8f c = {};
#pragma unroll
        for (int i = 0; i < 4; ++i) c = wmma_bf16(a[i], load_tile(brow, i * 32, g), c);
        int n = nt * 16 + lr;
#pragma unroll
        for (int r = 0; r < 8; ++r) {
            int m = m0 + r + 8 * g;
            float v = c[r];
            if (n < 16) {
                Bbuf[(long)m * 16 + n] = v;
            } else if (n < 32) {
                Cbuf[(long)m * 16 + (n - 16)] = v;
            } else {
                int d = n - 32;
                float u = v + bdt[d];
                float sp = (u > 20.f) ? u : __logf(1.f + __expf(u));   // softplus
                dtb[(long)m * 128 + d] = sp;
            }
        }
    }
}

// ---------------- selective scan: 1 block / sequence, 1 thread / channel ----------------
__global__ void __launch_bounds__(128) scan_kernel(
    const unsigned short* __restrict__ xp,
    const float* __restrict__ dtb,
    const float* __restrict__ Bbuf, const float* __restrict__ Cbuf,
    const unsigned short* __restrict__ sz,
    const float* __restrict__ Amat, const float* __restrict__ Dp,
    unsigned short* __restrict__ yg)
{
    __shared__ float Bs[DSTATE], Cs[DSTATE];
    int seq = blockIdx.x;
    int d   = threadIdx.x;
    float Ar[DSTATE];
#pragma unroll
    for (int s = 0; s < DSTATE; ++s) Ar[s] = Amat[d * DSTATE + s];
    float Dv = Dp[d];
    float h[DSTATE];
#pragma unroll
    for (int s = 0; s < DSTATE; ++s) h[s] = 0.f;
    int tok0 = seq * LBAND;
    for (int l = 0; l < LBAND; ++l) {
        int tk = tok0 + l;
        __syncthreads();
        if (threadIdx.x < DSTATE)           Bs[threadIdx.x] = Bbuf[(long)tk * DSTATE + threadIdx.x];
        else if (threadIdx.x < 2 * DSTATE)  Cs[threadIdx.x - DSTATE] = Cbuf[(long)tk * DSTATE + threadIdx.x - DSTATE];
        __syncthreads();
        float dt = dtb[(long)tk * 128 + d];
        float xv = bf2f(xp[(long)tk * 128 + d]);
        float y  = 0.f;
#pragma unroll
        for (int s = 0; s < DSTATE; ++s) {
            float dA = __expf(dt * Ar[s]);
            h[s] = dA * h[s] + (dt * Bs[s]) * xv;
            y += h[s] * Cs[s];
        }
        float o = (y + xv * Dv) * bf2f(sz[(long)tk * 128 + d]);
        yg[(long)tk * 128 + d] = f2bf(o);
    }
}

// ---------------- final: yg_f@Wf + yg_b(flip)@Wb + bias + residual, transposed store ----
// 1 block / M-tile; both 128x128 folded matrices (64KB) staged in LDS; 8 wmma/wave.
__global__ void __launch_bounds__(256) gemm_final(
    const unsigned short* __restrict__ ygf,
    const unsigned short* __restrict__ ygb,
    const unsigned short* __restrict__ wcT,     // [2][128*128]
    const float* __restrict__ biasTot,
    const float* __restrict__ x, float* __restrict__ out)
{
    __shared__ __align__(16) unsigned short lw[2 * 128 * 128];
    stage_to_lds(wcT, lw, 2 * 128 * 128 * 2);

    int wv = threadIdx.x >> 5, lane = threadIdx.x & 31;
    int g = lane >> 4, lr = lane & 15;
    int m0 = blockIdx.x * 16;
    int n0 = wv * 16;

    const unsigned short* arf = ygf + (long)(m0 + lr) * 128;
    const unsigned short* arb = ygb + (long)maprow(m0 + lr, 1) * 128;
    v16bf af[4], ab[4];
#pragma unroll
    for (int i = 0; i < 4; ++i) { af[i] = load_tile(arf, i * 32, g); ab[i] = load_tile(arb, i * 32, g); }

    const unsigned short* bfr = lw + (n0 + lr) * 128;
    const unsigned short* bbr = lw + 128 * 128 + (n0 + lr) * 128;
    v8f c = {};
#pragma unroll
    for (int i = 0; i < 4; ++i) c = wmma_bf16(af[i], load_tile(bfr, i * 32, g), c);
#pragma unroll
    for (int i = 0; i < 4; ++i) c = wmma_bf16(ab[i], load_tile(bbr, i * 32, g), c);

    int n = n0 + lr;
    float bias = biasTot[n];
#pragma unroll
    for (int r = 0; r < 8; ++r) {
        int m = m0 + r + 8 * g;          // token index
        int k  = m % LBAND;
        int bt = m / LBAND;
        int t  = bt % TT_;
        int b  = bt / TT_;
        long idx = (((long)b * CH + n) * TT_ + t) * LBAND + k;
        out[idx] = c[r] + bias + x[idx];
    }
}

// ---------------- launch ----------------
extern "C" void kernel_launch(void* const* d_in, const int* in_sizes, int n_in,
                              void* d_out, int out_size, void* d_ws, size_t ws_size,
                              hipStream_t stream)
{
    (void)in_sizes; (void)n_in; (void)out_size; (void)ws_size;
    const float* x     = (const float*)d_in[0];
    const float* ln_g  = (const float*)d_in[1];
    const float* ln_b  = (const float*)d_in[2];
    // per-direction params: f_ at 3..13, b_ at 14..24
    const float* Win[2]  = { (const float*)d_in[3],  (const float*)d_in[14] };
    const float* bin[2]  = { (const float*)d_in[4],  (const float*)d_in[15] };
    const float* cvw[2]  = { (const float*)d_in[5],  (const float*)d_in[16] };
    const float* cvb[2]  = { (const float*)d_in[6],  (const float*)d_in[17] };
    const float* Wx[2]   = { (const float*)d_in[7],  (const float*)d_in[18] };
    const float* Wdt[2]  = { (const float*)d_in[8],  (const float*)d_in[19] };
    const float* bdt[2]  = { (const float*)d_in[9],  (const float*)d_in[20] };
    const float* Alog[2] = { (const float*)d_in[10], (const float*)d_in[21] };
    const float* Dp[2]   = { (const float*)d_in[11], (const float*)d_in[22] };
    const float* Wout[2] = { (const float*)d_in[12], (const float*)d_in[23] };
    const float* bout[2] = { (const float*)d_in[13], (const float*)d_in[24] };
    const float* comb_W  = (const float*)d_in[25];
    const float* comb_b  = (const float*)d_in[26];
    float* out = (float*)d_out;

    char* ws = (char*)d_ws;
    size_t off = 0;
    auto alloc = [&](size_t bytes) -> void* {
        void* p = ws + off;
        off += (bytes + 255) & ~(size_t)255;
        return p;
    };
    unsigned short* XN  = (unsigned short*)alloc((size_t)TOK * 128 * 2);
    unsigned short* XZX = (unsigned short*)alloc((size_t)TOK * 128 * 2);
    unsigned short* SZ  = (unsigned short*)alloc((size_t)TOK * 128 * 2);
    unsigned short* XP  = (unsigned short*)alloc((size_t)TOK * 128 * 2);
    float*          DT  = (float*)alloc((size_t)TOK * 128 * 4);
    float*          BBf = (float*)alloc((size_t)TOK * DSTATE * 4);
    float*          CCf = (float*)alloc((size_t)TOK * DSTATE * 4);
    unsigned short* YG[2];
    YG[0] = (unsigned short*)alloc((size_t)TOK * 128 * 2);
    YG[1] = (unsigned short*)alloc((size_t)TOK * 128 * 2);
    unsigned short* WinT   = (unsigned short*)alloc((size_t)2 * 256 * 128 * 2);
    unsigned short* WeffT  = (unsigned short*)alloc((size_t)2 * 160 * 128 * 2);
    unsigned short* WcombT = (unsigned short*)alloc((size_t)2 * 128 * 128 * 2);
    float*          Amat   = (float*)alloc((size_t)2 * 128 * DSTATE * 4);
    float*          biasT  = (float*)alloc((size_t)128 * 4);

    setup_weights<<<256, 128, 0, stream>>>(
        Win[0], Wx[0], Wdt[0], Alog[0], Wout[0], bout[0],
        Win[1], Wx[1], Wdt[1], Alog[1], Wout[1], bout[1],
        comb_W, comb_b, WinT, WeffT, WcombT, Amat, biasT);

    ln_kernel<<<TOK / 8, 256, 0, stream>>>(x, ln_g, ln_b, XN);

    for (int dir = 0; dir < 2; ++dir) {
        gemm_in<<<MTILES, 256, 0, stream>>>(
            XN, WinT + dir * 256 * 128, bin[dir], XZX, SZ, dir);
        conv_silu<<<SEQN * 128 / 256, 256, 0, stream>>>(XZX, cvw[dir], cvb[dir], XP);
        gemm_x<<<MTILES, 256, 0, stream>>>(
            XP, WeffT + dir * 160 * 128, bdt[dir], BBf, CCf, DT);
        scan_kernel<<<SEQN, 128, 0, stream>>>(
            XP, DT, BBf, CCf, SZ, Amat + dir * 128 * DSTATE, Dp[dir], YG[dir]);
    }

    gemm_final<<<MTILES, 256, 0, stream>>>(
        YG[0], YG[1], WcombT, biasT, x, out);
}
